// PrediNet_6090263626344
// MI455X (gfx1250) — matrix operand order repaired and checked by reading, not compiled, not run on hardware
//
#include <hip/hip_runtime.h>
#include <math.h>

#define BGRAPHS 128
#define NNODES  2048
#define DDIM    256
#define KDIM    32
#define HP      32          // 2 head-sets * 16 heads
#define CHUNK   64          // rows of x staged per iteration
#define XS      264         // bf16 row stride for staged x (256 + 8 pad)
#define LEAKY_SLOPE 0.01f

typedef __attribute__((ext_vector_type(16))) __bf16       v16bf;
typedef __attribute__((ext_vector_type(8)))  float        v8f;
typedef __attribute__((ext_vector_type(4)))  unsigned int u32x4;
typedef __attribute__((ext_vector_type(4)))  float        f32x4;

union Frag { v16bf v; __bf16 b[16]; u32x4 u[2]; };

__device__ __forceinline__ unsigned short f2bf_bits(float f) {
    unsigned int u = __float_as_uint(f);
    return (unsigned short)((u + 0x7FFFu + ((u >> 16) & 1u)) >> 16);  // RNE
}
__device__ __forceinline__ __bf16 bits2bf(unsigned short s) {
    union { unsigned short s; __bf16 b; } u; u.s = s; return u.b;
}
__device__ __forceinline__ void cvt4(Frag& fr, int e, f32x4 f) {
    fr.b[e + 0] = bits2bf(f2bf_bits(f.x));
    fr.b[e + 1] = bits2bf(f2bf_bits(f.y));
    fr.b[e + 2] = bits2bf(f2bf_bits(f.z));
    fr.b[e + 3] = bits2bf(f2bf_bits(f.w));
}

// ---- dynamic-LDS byte offsets (total 318720 B <= 320 KB/WGP) ----
#define OFF_LOGITS 0         // [32][2048] f32 = 262144
#define OFF_XT     262144    // [64][264] bf16(ushort) = 33792
#define OFF_QDT    295936    // [32][256] bf16(ushort) = 16384 (hp, d)
#define OFF_POOL   312320    // [256] f32
#define OFF_Q      313344    // [32][32] f32 (hp, k)
#define OFF_RED    317440    // [32][8] f32
#define OFF_MAX    318464    // [32] f32
#define OFF_SUM    318592    // [32] f32
#define LDS_BYTES  318720
// overlays (regions dead by the time these are used):
#define OFF_F      OFF_XT        // [32][256] f32 = 32768  (<= 33792)
#define OFF_OUTH   OFF_QDT       // [32][16] f32 = 2048
#define OFF_DX     (OFF_QDT + 2048)  // [256] f32

extern "C" __global__ void __launch_bounds__(256)
predinet_fused_kernel(const float* __restrict__ inp,
                      const int*   __restrict__ batch_ids,
                      const float* __restrict__ Wk,
                      const float* __restrict__ Wq0,
                      const float* __restrict__ Wq1,
                      const float* __restrict__ We,
                      const float* __restrict__ Wout,
                      const float* __restrict__ bout,
                      float* __restrict__ out)
{
    (void)batch_ids;  // equal-size graphs: node n belongs to graph n / NNODES
    extern __shared__ char smem[];
    float*          logits = (float*)(smem + OFF_LOGITS);
    unsigned short* xt     = (unsigned short*)(smem + OFF_XT);
    unsigned short* qdt    = (unsigned short*)(smem + OFF_QDT);
    float*          pool   = (float*)(smem + OFF_POOL);
    float*          qbuf   = (float*)(smem + OFF_Q);
    float*          red    = (float*)(smem + OFF_RED);
    float*          maxv   = (float*)(smem + OFF_MAX);
    float*          sinv   = (float*)(smem + OFF_SUM);
    float*          F      = (float*)(smem + OFF_F);
    float*          outh   = (float*)(smem + OFF_OUTH);
    float*          dx     = (float*)(smem + OFF_DX);

    const int tid  = threadIdx.x;
    const int lane = tid & 31;
    const int wid  = tid >> 5;           // 8 waves
    const int b    = blockIdx.x;
    const float* xg = inp + (size_t)b * NNODES * DDIM;

    // prefetch address pattern: 256 threads cover a 64-row (64 KB) chunk at 256-B spacing
    const int pf_row = tid >> 2;
    const int pf_col = (tid & 3) * 64;

    // ---------- pass 1: per-graph column max pool (coalesced stream) ----------
    {
        float m = -INFINITY;
        for (int n = 0; n < NNODES; ++n)
            m = fmaxf(m, xg[(size_t)n * DDIM + tid]);
        pool[tid] = m;
    }
    __syncthreads();

    // ---------- q[hp][k] = pooled . Wq[hp*32+k][:]  (1024 dots) ----------
    for (int i = 0; i < 4; ++i) {
        int e   = tid + 256 * i;               // 0..1023
        int hs  = e >> 9;                      // head-set
        int row = e & 511;                     // h*32 + k
        const float* wrow = (hs ? Wq1 : Wq0) + (size_t)row * DDIM;
        float acc = 0.f;
        for (int d = 0; d < DDIM; ++d) acc += pool[d] * wrow[d];
        qbuf[e] = acc;                         // flat index == hp*32 + k
    }
    __syncthreads();

    // ---------- qdT[d][hp] = sum_k q[hp][k] * Wk[k][d]  (fold Wk into q) ----------
    {
        const int d = tid;
        for (int hp = 0; hp < HP; ++hp) {
            const float* q = qbuf + hp * KDIM;
            float acc = 0.f;
            for (int k = 0; k < KDIM; ++k) acc += q[k] * Wk[(size_t)k * DDIM + d];
            qdt[hp * DDIM + d] = f2bf_bits(acc);
        }
    }
    __syncthreads();

    // ---------- pass 2: logits[hp][n] = x @ qdT  (bf16 WMMA) ----------
    for (int c = 0; c < NNODES / CHUNK; ++c) {
        const int base = c * CHUNK;
        __syncthreads();  // previous chunk's readers done before restaging
        for (int r2 = 0; r2 < CHUNK; ++r2)
            xt[r2 * XS + tid] = f2bf_bits(xg[(size_t)(base + r2) * DDIM + tid]);
        if (c + 1 < NNODES / CHUNK)           // pull next chunk toward the WGP
            __builtin_prefetch(&xg[(size_t)(base + CHUNK + pf_row) * DDIM + pf_col], 0, 3);
        __syncthreads();

        const int mt = wid >> 1, nt = wid & 1;       // 4 M-tiles x 2 N-tiles = 8 waves
        v8f acc = {};
        const int arow = mt * 16 + (lane & 15);
        const int dsel = ((lane >> 4) & 1) * 8;      // A layout half-select
        for (int kk = 0; kk < DDIM / 32; ++kk) {
            Frag a, bm;
            const int d0 = kk * 32 + dsel;
            a.u[0] = *(const u32x4*)&xt[arow * XS + d0];        // K runs d0..d0+7
            a.u[1] = *(const u32x4*)&xt[arow * XS + d0 + 16];   // and d0+16..d0+23
            const int bn = nt * 16 + (lane & 15);               // B: N = lane&15, K as A
            bm.u[0] = *(const u32x4*)&qdt[bn * DDIM + d0];
            bm.u[1] = *(const u32x4*)&qdt[bn * DDIM + d0 + 16];
            acc = __builtin_amdgcn_wmma_f32_16x16x32_bf16(
                      false, a.v, false, bm.v, (short)0, acc, false, false);
        }
        const int hh = nt * 16 + (lane & 15);                   // C: N = lane&15
        const int nb = base + mt * 16 + ((lane >> 4) & 1) * 8;  // C: M = r + 8*(lane>=16)
#pragma unroll
        for (int r = 0; r < 8; ++r)
            logits[hh * NNODES + nb + r] = acc[r];
    }
    __syncthreads();

    // ---------- softmax over n per combined head (keep unnormalized exp) ----------
    {
        const int g = tid >> 3, sub = tid & 7;   // 8 threads per head
        float m = -INFINITY;
        for (int n = sub; n < NNODES; n += 8)
            m = fmaxf(m, logits[g * NNODES + n]);
        red[g * 8 + sub] = m;
        __syncthreads();
        if (tid < HP) {
            float mm = red[tid * 8];
            for (int i = 1; i < 8; ++i) mm = fmaxf(mm, red[tid * 8 + i]);
            maxv[tid] = mm;
        }
        __syncthreads();
        const float mh = maxv[g];
        float s = 0.f;
        for (int n = sub; n < NNODES; n += 8) {
            float e = __expf(logits[g * NNODES + n] - mh);
            logits[g * NNODES + n] = e;       // att (unnormalized), in place
            s += e;
        }
        red[g * 8 + sub] = s;
        __syncthreads();
        if (tid < HP) {
            float ss = 0.f;
            for (int i = 0; i < 8; ++i) ss += red[tid * 8 + i];
            sinv[tid] = 1.0f / ss;
        }
    }
    __syncthreads();

    // ---------- pass 3: F[hp][d] = att @ x  (bf16 WMMA, persistent C) ----------
    v8f facc[4] = {v8f{}, v8f{}, v8f{}, v8f{}};
    const int nt0 = wid * 2;   // each wave owns d-tiles {2w, 2w+1} x head-tiles {0,1}
    for (int c = 0; c < NNODES / CHUNK; ++c) {
        const int base = c * CHUNK;
        __syncthreads();
        for (int r2 = 0; r2 < CHUNK; ++r2)
            xt[r2 * XS + tid] = f2bf_bits(xg[(size_t)(base + r2) * DDIM + tid]);
        if (c + 1 < NNODES / CHUNK)
            __builtin_prefetch(&xg[(size_t)(base + CHUNK + pf_row) * DDIM + pf_col], 0, 3);
        __syncthreads();

        for (int ks = 0; ks < CHUNK / 32; ++ks) {
            const int n0 = base + ks * 32;
            Frag afr[2];
#pragma unroll
            for (int mt = 0; mt < 2; ++mt) {     // A = att rows (heads), K = n
                const int hh = mt * 16 + (lane & 15);
                const float* src = &logits[hh * NNODES + n0 + ((lane >> 4) & 1) * 8];
                cvt4(afr[mt], 0,  *(const f32x4*)(src));
                cvt4(afr[mt], 4,  *(const f32x4*)(src + 4));
                cvt4(afr[mt], 8,  *(const f32x4*)(src + 16));
                cvt4(afr[mt], 12, *(const f32x4*)(src + 20));
            }
            Frag bfr[2];
#pragma unroll
            for (int j = 0; j < 2; ++j) {        // B = x, N = d = lane&15, K = n in table order
                const unsigned short* p = &xt[(ks * 32 + ((lane >> 4) & 1) * 8) * XS + (nt0 + j) * 16 + (lane & 15)];
                for (int s = 0; s < 16; ++s)
                    bfr[j].b[s] = bits2bf(p[(s < 8 ? s : s + 8) * XS]);
            }
#pragma unroll
            for (int mt = 0; mt < 2; ++mt)
#pragma unroll
                for (int j = 0; j < 2; ++j)
                    facc[mt * 2 + j] = __builtin_amdgcn_wmma_f32_16x16x32_bf16(
                        false, afr[mt].v, false, bfr[j].v, (short)0,
                        facc[mt * 2 + j], false, false);
        }
    }
    __syncthreads();   // xt dead; F overlays it

    // scale by 1/sum and write feat to LDS
#pragma unroll
    for (int mt = 0; mt < 2; ++mt)
#pragma unroll
        for (int j = 0; j < 2; ++j) {
            const int dcol = (nt0 + j) * 16 + (lane & 15);
#pragma unroll
            for (int r = 0; r < 8; ++r) {
                const int hh = mt * 16 + r + 8 * ((lane >> 4) & 1);
                F[hh * DDIM + dcol] = facc[mt * 2 + j][r] * sinv[hh];
            }
        }
    __syncthreads();

    // ---------- outh[hp][r] = F[hp][:] . We[r][:] ----------
    for (int i = 0; i < 2; ++i) {
        int e = tid + 256 * i;                 // 0..511
        int hh = e >> 4, r = e & 15;
        const float* we = We + (size_t)r * DDIM;
        const float* fr = F + hh * DDIM;
        float acc = 0.f;
        for (int d = 0; d < DDIM; ++d) acc += fr[d] * we[d];
        outh[e] = acc;
    }
    __syncthreads();
    dx[tid] = outh[tid] - outh[tid + 256];     // head-set 0 minus head-set 1
    __syncthreads();

    // ---------- out = dx @ Wout.T + bout, LeakyReLU ----------
    {
        const float* wrow = Wout + (size_t)tid * DDIM;   // Wout [256][256]
        float acc = bout[tid];
        for (int cc = 0; cc < DDIM; ++cc) acc += dx[cc] * wrow[cc];
        out[(size_t)b * DDIM + tid] = (acc >= 0.f) ? acc : LEAKY_SLOPE * acc;
    }
}

extern "C" void kernel_launch(void* const* d_in, const int* in_sizes, int n_in,
                              void* d_out, int out_size, void* d_ws, size_t ws_size,
                              hipStream_t stream) {
    (void)in_sizes; (void)n_in; (void)out_size; (void)d_ws; (void)ws_size;
    const float* inp  = (const float*)d_in[0];
    const int*   bid  = (const int*)d_in[1];
    const float* Wk   = (const float*)d_in[2];
    const float* Wq0  = (const float*)d_in[3];
    const float* Wq1  = (const float*)d_in[4];
    const float* We   = (const float*)d_in[5];
    const float* Wout = (const float*)d_in[6];
    const float* bout = (const float*)d_in[7];

    // allow >default dynamic LDS (gfx1250: 320 KB per WGP)
    hipFuncSetAttribute(reinterpret_cast<const void*>(predinet_fused_kernel),
                        hipFuncAttributeMaxDynamicSharedMemorySize, LDS_BYTES);

    predinet_fused_kernel<<<dim3(BGRAPHS), dim3(256), LDS_BYTES, stream>>>(
        inp, bid, Wk, Wq0, Wq1, We, Wout, bout, (float*)d_out);
}
